// CrossModalAttention_18356690223819
// MI455X (gfx1250) — matrix operand order repaired and checked by reading, not compiled, not run on hardware
//
#include <hip/hip_runtime.h>

// CrossModalAttention on MI455X (gfx1250, wave32).
//
// softmax over one key == 1 -> attn == V; fold GEMMs: out = kv @ (Wo@Wv).T + (Wo@bv+bo),
// fused residual+LayerNorm epilogue. Arithmetic intensity ~254 FLOP/B -> compute-bound,
// so run everything on V_WMMA_F32_16X16X32_BF16 using the bf16x3 split-precision scheme
// (hi*hi + hi*lo + lo*hi, f32 accumulate) for ~fp32 accuracy at bf16 matrix throughput.

#define D_DIM 1024
#define LN_EPS 1e-5f

typedef float v8f  __attribute__((ext_vector_type(8)));
typedef __bf16 v8bf  __attribute__((ext_vector_type(8)));
typedef __bf16 v16bf __attribute__((ext_vector_type(16)));

// split fp32 -> bf16 hi (truncated top16, exact) + bf16 lo (residual, RNE)
__device__ __forceinline__ void split2(float v, __bf16& h, __bf16& l) {
  unsigned u = __float_as_uint(v) & 0xFFFF0000u;
  float hf = __uint_as_float(u);
  h = (__bf16)hf;          // exact: low mantissa bits are zero
  l = (__bf16)(v - hf);    // exact residual in f32, rounded to bf16
}

// A-fragment (16-bit 16x32 layout): per lane, 16 contiguous elems
__device__ __forceinline__ v16bf load_frag_a(const __bf16* p) {
  v8bf x0 = *(const v8bf*)(p);
  v8bf x1 = *(const v8bf*)(p + 8);
  return __builtin_shufflevector(x0, x1, 0,1,2,3,4,5,6,7,8,9,10,11,12,13,14,15);
}
// B-fragment (16-bit 32x16 layout): per lane, 16 contiguous elems
__device__ __forceinline__ v16bf load_frag_b(const __bf16* p) {
  v8bf x0 = *(const v8bf*)(p);
  v8bf x1 = *(const v8bf*)(p + 8);
  return __builtin_shufflevector(x0, x1, 0,1,2,3,4,5,6,7,8,9,10,11,12,13,14,15);
}

__device__ __forceinline__ v8f wmma3(v16bf ahi, v16bf alo, v16bf bhi, v16bf blo, v8f c) {
  c = __builtin_amdgcn_wmma_f32_16x16x32_bf16(false, ahi, false, bhi, (short)0, c, false, false);
  c = __builtin_amdgcn_wmma_f32_16x16x32_bf16(false, ahi, false, blo, (short)0, c, false, false);
  c = __builtin_amdgcn_wmma_f32_16x16x32_bf16(false, alo, false, bhi, (short)0, c, false, false);
  return c;
}

// ---------------- split Wo (row-major, no transpose) ----------------
__global__ __launch_bounds__(256)
void split_kernel(const float* __restrict__ x, __bf16* __restrict__ hi,
                  __bf16* __restrict__ lo, int n) {
  const int i = blockIdx.x * 256 + threadIdx.x;
  if (i >= n) return;
  split2(x[i], hi[i], lo[i]);
}

// ---------------- split + transpose Wv: [j][k] fp32 -> [k][j] bf16 hi/lo ----------------
__global__ __launch_bounds__(256)
void split_transpose_kernel(const float* __restrict__ w,
                            __bf16* __restrict__ thi, __bf16* __restrict__ tlo) {
  __shared__ float tile[32][33];
  const int bx = blockIdx.x * 32;               // input col base
  const int by = blockIdx.y * 32;               // input row base
  const int tx = threadIdx.x & 31, ty = threadIdx.x >> 5;   // 32x8
#pragma unroll
  for (int i = 0; i < 32; i += 8)
    tile[ty + i][tx] = w[(size_t)(by + ty + i) * D_DIM + bx + tx];
  __syncthreads();
#pragma unroll
  for (int i = 0; i < 32; i += 8) {
    const float v = tile[tx][ty + i];           // = w[by+tx][bx+ty+i]
    const size_t o = (size_t)(bx + ty + i) * D_DIM + by + tx;
    split2(v, thi[o], tlo[o]);
  }
}

// ---------------- fused weight GEMM: M = Wo @ Wv  via  Wo @ (Wvt).T ----------------
// Output row-major [n][k] as bf16 hi/lo. Block: 8 waves, 32 n-rows x 1024 k-cols.
__global__ __launch_bounds__(256, 1)
void fuse_gemm_kernel(const __bf16* __restrict__ WoHi, const __bf16* __restrict__ WoLo,
                      const __bf16* __restrict__ WvtHi, const __bf16* __restrict__ WvtLo,
                      __bf16* __restrict__ MHi, __bf16* __restrict__ MLo) {
  const int n0    = blockIdx.x * 32;
  const int tid   = threadIdx.x;
  const int wave  = tid >> 5, lane = tid & 31;
  const int rg    = wave >> 2;        // n row-group 0..1
  const int cg    = wave & 3;         // k col-group 0..3
  const int lrow  = lane & 15;
  const int khalf = lane >> 4;

  const __bf16* aHi = WoHi + (size_t)(n0 + rg * 16 + lrow) * D_DIM + khalf * 16;
  const __bf16* aLo = WoLo + (size_t)(n0 + rg * 16 + lrow) * D_DIM + khalf * 16;
  const size_t brow = (size_t)(cg * 256 + lrow) * D_DIM;

  v8f acc[16];
  const v8f vz = {0.f,0.f,0.f,0.f,0.f,0.f,0.f,0.f};
#pragma unroll
  for (int t = 0; t < 16; ++t) acc[t] = vz;

  for (int j = 0; j < D_DIM; j += 32) {
    v16bf ahi = load_frag_a(aHi + j);
    v16bf alo = load_frag_a(aLo + j);
    const int jb = j + khalf * 16;
#pragma unroll
    for (int t = 0; t < 16; ++t) {
      const size_t off = brow + (size_t)t * 16 * D_DIM + jb;
      v16bf bhi = load_frag_b(WvtHi + off);
      v16bf blo = load_frag_b(WvtLo + off);
      acc[t] = wmma3(ahi, alo, bhi, blo, acc[t]);
    }
  }
  // store split hi/lo, layout M[n][k]
  const int rbase = n0 + rg * 16 + khalf * 8;
#pragma unroll
  for (int i = 0; i < 8; ++i) {
#pragma unroll
    for (int t = 0; t < 16; ++t) {
      const size_t o = (size_t)(rbase + i) * D_DIM + cg * 256 + t * 16 + lrow;
      split2(acc[t][i], MHi[o], MLo[o]);
    }
  }
}

// ---------------- fused bias: c = Wo @ bv + bo (tiny, fp32) ----------------
__global__ __launch_bounds__(256)
void fuse_bias_kernel(const float* __restrict__ Wo, const float* __restrict__ bv,
                      const float* __restrict__ bo, float* __restrict__ c) {
  const int n = blockIdx.x * blockDim.x + threadIdx.x;
  if (n >= D_DIM) return;
  float s = bo[n];
  for (int j = 0; j < D_DIM; ++j)
    s = fmaf(Wo[(size_t)n * D_DIM + j], bv[j], s);
  c[n] = s;
}

// ---------------- main: GEMM + bias + residual + LayerNorm ----------------
// Block: 256 threads = 8 waves (2 row-groups x 4 col-groups), 32 rows x 1024 cols.
// A (X_kv) is split fp32->bf16 hi/lo into LDS per 256-wide K chunk; B streams from
// L2-resident M hi/lo.
__global__ __launch_bounds__(256, 1)
void attn_ln_kernel(const float* __restrict__ Ximg, const float* __restrict__ Xtxt,
                    const __bf16* __restrict__ MHi1, const __bf16* __restrict__ MLo1,
                    const __bf16* __restrict__ MHi2, const __bf16* __restrict__ MLo2,
                    const float* __restrict__ c1, const float* __restrict__ c2,
                    const float* __restrict__ g1, const float* __restrict__ b1,
                    const float* __restrict__ g2, const float* __restrict__ b2,
                    float* __restrict__ out, int Bn) {
  const bool s0 = (blockIdx.y == 0);
  const float* __restrict__ Xkv = s0 ? Xtxt : Ximg;
  const float* __restrict__ Xq  = s0 ? Ximg : Xtxt;
  const __bf16* __restrict__ MwHi = s0 ? MHi1 : MHi2;
  const __bf16* __restrict__ MwLo = s0 ? MLo1 : MLo2;
  const float* __restrict__ cb = s0 ? c1 : c2;
  const float* __restrict__ ga = s0 ? g1 : g2;
  const float* __restrict__ be = s0 ? b1 : b2;
  float* __restrict__ outp = out + (s0 ? (size_t)0 : (size_t)Bn * D_DIM);

  __shared__ __bf16 sAhi[32][264];   // 256-wide K chunk + 8-elem pad (bank-conflict free)
  __shared__ __bf16 sAlo[32][264];
  __shared__ float redS[4][32];
  __shared__ float redQ[4][32];
  __shared__ float rowMu[32];
  __shared__ float rowRs[32];

  const int r0    = blockIdx.x * 32;
  const int tid   = threadIdx.x;
  const int wave  = tid >> 5, lane = tid & 31;
  const int rg    = wave >> 2;
  const int cg    = wave & 3;
  const int lrow  = lane & 15;
  const int khalf = lane >> 4;
  const int arow  = rg * 16 + lrow;
  const size_t brow = (size_t)(cg * 256 + lrow) * D_DIM;

  v8f acc[16];
  const v8f vz = {0.f,0.f,0.f,0.f,0.f,0.f,0.f,0.f};
#pragma unroll
  for (int t = 0; t < 16; ++t) acc[t] = vz;

  for (int k0 = 0; k0 < D_DIM; k0 += 256) {
    if (k0) __syncthreads();
    // cooperative load + split of the 32x256 X_kv chunk
#pragma unroll 4
    for (int r = 0; r < 32; ++r) {
      const float v = Xkv[(size_t)(r0 + r) * D_DIM + k0 + tid];
      split2(v, sAhi[r][tid], sAlo[r][tid]);
    }
    __syncthreads();

    for (int kk = 0; kk < 256; kk += 32) {
      v16bf ahi = load_frag_a(&sAhi[arow][kk + khalf * 16]);
      v16bf alo = load_frag_a(&sAlo[arow][kk + khalf * 16]);
      const int kb = k0 + kk + khalf * 16;
#pragma unroll
      for (int t = 0; t < 16; ++t) {
        const size_t off = brow + (size_t)t * 16 * D_DIM + kb;
        v16bf bhi = load_frag_b(MwHi + off);
        v16bf blo = load_frag_b(MwLo + off);
        acc[t] = wmma3(ahi, alo, bhi, blo, acc[t]);
      }
    }
  }

  // ---- epilogue: + fused bias + residual ----
  float cbv[16];
#pragma unroll
  for (int t = 0; t < 16; ++t) cbv[t] = cb[cg * 256 + t * 16 + lrow];

  const int rbase = r0 + rg * 16 + khalf * 8;   // C layout: VGPR i -> row i (+8 for hi lanes)
#pragma unroll
  for (int i = 0; i < 8; ++i) {
    const float* xq = Xq + (size_t)(rbase + i) * D_DIM;
#pragma unroll
    for (int t = 0; t < 16; ++t)
      acc[t][i] += cbv[t] + xq[cg * 256 + t * 16 + lrow];
  }

  // ---- LayerNorm ----
#pragma unroll
  for (int i = 0; i < 8; ++i) {
    float s = 0.f, q = 0.f;
#pragma unroll
    for (int t = 0; t < 16; ++t) { const float v = acc[t][i]; s += v; q = fmaf(v, v, q); }
#pragma unroll
    for (int off = 1; off < 16; off <<= 1) {   // xor<16 stays within each half-wave
      s += __shfl_xor(s, off, 32);
      q += __shfl_xor(q, off, 32);
    }
    if (lrow == 0) {
      const int rloc = rg * 16 + khalf * 8 + i;
      redS[cg][rloc] = s;
      redQ[cg][rloc] = q;
    }
  }
  __syncthreads();
  if (tid < 32) {
    const float s = redS[0][tid] + redS[1][tid] + redS[2][tid] + redS[3][tid];
    const float q = redQ[0][tid] + redQ[1][tid] + redQ[2][tid] + redQ[3][tid];
    const float mu  = s * (1.f / D_DIM);
    const float var = q * (1.f / D_DIM) - mu * mu;
    rowMu[tid] = mu;
    rowRs[tid] = rsqrtf(var + LN_EPS);
  }
  __syncthreads();

  float gv[16], bv[16];
#pragma unroll
  for (int t = 0; t < 16; ++t) {
    const int n = cg * 256 + t * 16 + lrow;
    gv[t] = ga[n];
    bv[t] = be[n];
  }
#pragma unroll
  for (int i = 0; i < 8; ++i) {
    const int rloc = rg * 16 + khalf * 8 + i;
    const float mu = rowMu[rloc], rs = rowRs[rloc];
    float* orow = outp + (size_t)(r0 + rloc) * D_DIM;
#pragma unroll
    for (int t = 0; t < 16; ++t) {
      const float v = (acc[t][i] - mu) * rs * gv[t] + bv[t];
      __builtin_nontemporal_store(v, orow + cg * 256 + t * 16 + lrow);
    }
  }
}

extern "C" void kernel_launch(void* const* d_in, const int* in_sizes, int n_in,
                              void* d_out, int out_size, void* d_ws, size_t ws_size,
                              hipStream_t stream) {
  const float* img    = (const float*)d_in[0];
  const float* txt    = (const float*)d_in[1];
  const float* in_w1  = (const float*)d_in[2];
  const float* in_b1  = (const float*)d_in[3];
  const float* out_w1 = (const float*)d_in[4];
  const float* out_b1 = (const float*)d_in[5];
  const float* in_w2  = (const float*)d_in[6];
  const float* in_b2  = (const float*)d_in[7];
  const float* out_w2 = (const float*)d_in[8];
  const float* out_b2 = (const float*)d_in[9];
  const float* g1     = (const float*)d_in[10];
  const float* b1     = (const float*)d_in[11];
  const float* g2     = (const float*)d_in[12];
  const float* b2     = (const float*)d_in[13];

  const int Bn = in_sizes[0] / D_DIM;          // 16384
  const size_t E = (size_t)D_DIM * D_DIM;      // 1M elements

  // workspace: 12 bf16 matrices (2MB each) + 2 fp32 bias vectors  => ~24MB
  __bf16* bp = (__bf16*)d_ws;
  __bf16 *MHi1 = bp,        *MLo1 = bp + E,    *MHi2 = bp + 2*E,  *MLo2 = bp + 3*E;
  __bf16 *WoHi1 = bp + 4*E, *WoLo1 = bp + 5*E, *WoHi2 = bp + 6*E, *WoLo2 = bp + 7*E;
  __bf16 *WvtHi1 = bp + 8*E,*WvtLo1 = bp + 9*E,*WvtHi2 = bp + 10*E,*WvtLo2 = bp + 11*E;
  float* c1 = (float*)(bp + 12*E);
  float* c2 = c1 + D_DIM;

  const size_t wvOff = 2 * E;                  // Wv = rows [2D,3D) of packed in_w
  const dim3 gT(D_DIM / 32, D_DIM / 32);

  split_kernel<<<dim3((int)(E / 256)), 256, 0, stream>>>(out_w1, WoHi1, WoLo1, (int)E);
  split_kernel<<<dim3((int)(E / 256)), 256, 0, stream>>>(out_w2, WoHi2, WoLo2, (int)E);
  split_transpose_kernel<<<gT, 256, 0, stream>>>(in_w1 + wvOff, WvtHi1, WvtLo1);
  split_transpose_kernel<<<gT, 256, 0, stream>>>(in_w2 + wvOff, WvtHi2, WvtLo2);

  fuse_gemm_kernel<<<dim3(D_DIM / 32), 256, 0, stream>>>(WoHi1, WoLo1, WvtHi1, WvtLo1, MHi1, MLo1);
  fuse_gemm_kernel<<<dim3(D_DIM / 32), 256, 0, stream>>>(WoHi2, WoLo2, WvtHi2, WvtLo2, MHi2, MLo2);
  fuse_bias_kernel<<<dim3(D_DIM / 256), 256, 0, stream>>>(out_w1, in_b1 + 2 * D_DIM, out_b1, c1);
  fuse_bias_kernel<<<dim3(D_DIM / 256), 256, 0, stream>>>(out_w2, in_b2 + 2 * D_DIM, out_b2, c2);

  dim3 gM(Bn / 32, 2);   // y: 0 = attended_image, 1 = attended_text
  attn_ln_kernel<<<gM, 256, 0, stream>>>(img, txt, MHi1, MLo1, MHi2, MLo2,
                                         c1, c2, g1, b1, g2, b2, (float*)d_out, Bn);
}
